// MultiHeadAttention_4458176053366
// MI455X (gfx1250) — matrix operand order repaired and checked by reading, not compiled, not run on hardware
//
#include <hip/hip_runtime.h>
#include <hip/hip_bf16.h>

#define EDIM  1024
#define NH    16
#define DH    64
#define BATCH 4
#define NSEQ  1024
#define QKSCALE 0.125f        // 64^-0.5
#define SSHIFT  8.0f          // static softmax shift; scores ~N(0,1), |s| << 8+126
#define LOG2E   1.4426950408889634f

#if __has_builtin(__builtin_amdgcn_exp2f)
#define EXP2F(x) __builtin_amdgcn_exp2f(x)
#else
#define EXP2F(x) exp2f(x)
#endif

typedef __attribute__((ext_vector_type(16))) _Float16 v16h;
typedef __attribute__((ext_vector_type(8)))  _Float16 v8h;
typedef __attribute__((ext_vector_type(8)))  float    v8f;

union V16U { v16h v; v8h h[2]; };

// A-matrix (16x32 f16) fragment, this code's own k order (not the ISA table order):
// lane lo = row M; halves 0..15 at K = hi*16+[0..16) (contiguous, matches B loads)
__device__ __forceinline__ v16h load_a_frag(const _Float16* rowChunk, int hi) {
    V16U u;
    u.h[0] = *(const v8h*)(rowChunk + hi * 16);
    u.h[1] = *(const v8h*)(rowChunk + hi * 16 + 8);
    return u.v;
}

__device__ __forceinline__ v8f wmma32(v16h a, v16h b, v8f c) {
    return __builtin_amdgcn_wmma_f32_16x16x32_f16(false, a, false, b, (short)0, c, false, false);
}

// ---------------- conversion kernels ----------------
__global__ void cvt_qkv_kernel(const float* __restrict__ q, const float* __restrict__ k,
                               const float* __restrict__ v,
                               _Float16* __restrict__ qh, _Float16* __restrict__ kh,
                               _Float16* __restrict__ vT) {
    size_t idx = (size_t)blockIdx.x * blockDim.x + threadIdx.x;   // over B*N*E
    int e = (int)(idx % EDIM);
    size_t bn = idx / EDIM;
    int n = (int)(bn % NSEQ);
    int b = (int)(bn / NSEQ);
    int h = e >> 6, d = e & 63;
    size_t qi = (((size_t)b * NH + h) * NSEQ + n) * DH + d;
    qh[qi] = (_Float16)q[idx];
    kh[qi] = (_Float16)k[idx];
    vT[(((size_t)b * NH + h) * DH + d) * NSEQ + n] = (_Float16)v[idx];
}

__global__ void cvt_w_kernel(const float* __restrict__ w, _Float16* __restrict__ wh) {
    size_t i = (size_t)blockIdx.x * blockDim.x + threadIdx.x;     // over E*E
    wh[i] = (_Float16)w[i];
}

// pack attn_mask (int32, B*N*N) into bitmask words: mbits[(b*N+q)*32 + w] bit t = mask[b,q,w*32+t]!=0
__global__ void cvt_mask_kernel(const int* __restrict__ m, unsigned int* __restrict__ mbits) {
    size_t w = (size_t)blockIdx.x * blockDim.x + threadIdx.x;     // over B*N*(N/32)
    const int* src = m + w * 32;
    unsigned int bits = 0;
    #pragma unroll
    for (int t = 0; t < 32; ++t) bits |= (src[t] != 0 ? 1u : 0u) << t;
    mbits[w] = bits;
}

// ---------------- fused attention ----------------
// one wave (32 threads) per (b, h, 32-row q tile); 2 accumulator sets share B fragments
__global__ __launch_bounds__(32)
void attn_kernel(const _Float16* __restrict__ qh, const _Float16* __restrict__ kh,
                 const _Float16* __restrict__ vT, const unsigned int* __restrict__ mbits,
                 const float* __restrict__ prev, _Float16* __restrict__ Oh) {
    __shared__ _Float16 lds[32 * 32];

    const int lane = threadIdx.x & 31;
    const int lo = lane & 15, hi = lane >> 4;

    const int tile = blockIdx.x;
    const int qt = tile & 31;          // N/32 = 32 double-tiles
    const int h  = (tile >> 5) & 15;
    const int b  = tile >> 9;
    const int q0 = qt * 32;

    const float kSc = QKSCALE * LOG2E;      // fold log2e into the WMMA-output scale
    const float kSh = -SSHIFT * LOG2E;      // fold log2e into the static shift

    const _Float16* qbase = qh + (((size_t)b * NH + h) * NSEQ) * DH;
    const _Float16* kbase = kh + (((size_t)b * NH + h) * NSEQ) * DH;
    const _Float16* vbase = vT + (((size_t)b * NH + h) * DH) * NSEQ;
    const unsigned int* mb = mbits + ((size_t)b * NSEQ + q0) * (NSEQ / 32);
    const float* pbase = prev + (((size_t)b * NH + h) * NSEQ + q0) * NSEQ;

    // Q fragments for both 16-row sub-tiles: head dim 64 -> 2 chunks of K=32
    v16h aq[2][2];
    #pragma unroll
    for (int qi = 0; qi < 2; ++qi) {
        const _Float16* qr = qbase + (size_t)(q0 + qi * 16 + lo) * DH;
        aq[qi][0] = load_a_frag(qr, hi);
        aq[qi][1] = load_a_frag(qr + 32, hi);
    }

    // per-LANE partial exp-sums (softmax with static shift; exact by shift invariance)
    float pl[2][8];
    #pragma unroll
    for (int qi = 0; qi < 2; ++qi)
        #pragma unroll
        for (int r = 0; r < 8; ++r) pl[qi][r] = 0.0f;

    // ---- pass 1: per-lane exp-sum over all key tiles ----
    for (int jj = 0; jj < NSEQ / 32; ++jj) {
        unsigned int mw[2][8];
        #pragma unroll
        for (int qi = 0; qi < 2; ++qi)
            #pragma unroll
            for (int r = 0; r < 8; ++r)
                mw[qi][r] = mb[(size_t)(qi * 16 + r + hi * 8) * (NSEQ / 32) + jj];
        #pragma unroll
        for (int jt = 0; jt < 2; ++jt) {
            const int j = jj * 2 + jt;
            const _Float16* kr = kbase + (size_t)(j * 16 + lo) * DH + hi * 16;
            v16h b0 = *(const v16h*)(kr);
            v16h b1 = *(const v16h*)(kr + 32);
            #pragma unroll
            for (int qi = 0; qi < 2; ++qi) {
                v8f c = {};
                c = wmma32(aq[qi][0], b0, c);
                c = wmma32(aq[qi][1], b1, c);
                #pragma unroll
                for (int r = 0; r < 8; ++r) {
                    float e = EXP2F(__builtin_fmaf(c[r], kSc, kSh));
                    unsigned int bit = (mw[qi][r] >> (jt * 16 + lo)) & 1u;
                    pl[qi][r] += bit ? e : 0.0f;
                }
            }
        }
    }

    // ---- cross-lane sum; fold -log2(l) into the per-row exponent shift ----
    float rsh[2][8];
    #pragma unroll
    for (int qi = 0; qi < 2; ++qi)
        #pragma unroll
        for (int r = 0; r < 8; ++r) {
            float l = pl[qi][r];
            l += __shfl_xor(l, 1);
            l += __shfl_xor(l, 2);
            l += __shfl_xor(l, 4);
            l += __shfl_xor(l, 8);
            rsh[qi][r] = kSh - __log2f(fmaxf(l, 1.0e-37f));   // exp2(y+rsh) = exp(s-8)/l
        }

    // ---- pass 2: P = softmax(S) + attn_prev, O += P * V ----
    v8f o[2][4];
    #pragma unroll
    for (int qi = 0; qi < 2; ++qi)
        #pragma unroll
        for (int dt = 0; dt < 4; ++dt) o[qi][dt] = (v8f){};

    for (int jj = 0; jj < NSEQ / 32; ++jj) {
        unsigned int mw[2][8];
        #pragma unroll
        for (int qi = 0; qi < 2; ++qi)
            #pragma unroll
            for (int r = 0; r < 8; ++r)
                mw[qi][r] = mb[(size_t)(qi * 16 + r + hi * 8) * (NSEQ / 32) + jj];
        #pragma unroll
        for (int jt = 0; jt < 2; ++jt) {
            const int j = jj * 2 + jt;
            const _Float16* kr = kbase + (size_t)(j * 16 + lo) * DH + hi * 16;
            v16h b0 = *(const v16h*)(kr);
            v16h b1 = *(const v16h*)(kr + 32);
            #pragma unroll
            for (int qi = 0; qi < 2; ++qi) {
                v8f c = {};
                c = wmma32(aq[qi][0], b0, c);
                c = wmma32(aq[qi][1], b1, c);
                #pragma unroll
                for (int r = 0; r < 8; ++r) {
                    float e = EXP2F(__builtin_fmaf(c[r], kSc, rsh[qi][r]));
                    unsigned int bit = (mw[qi][r] >> (jt * 16 + lo)) & 1u;
                    e = bit ? e : 0.0f;
                    float p = e + pbase[(size_t)(qi * 16 + r + hi * 8) * NSEQ + j * 16 + lo];
                    lds[(qi * 16 + r + hi * 8) * 32 + jt * 16 + lo] = (_Float16)p;
                }
            }
        }
        __syncthreads();
        v16h ap[2];
        #pragma unroll
        for (int qi = 0; qi < 2; ++qi)
            ap[qi] = load_a_frag(&lds[(qi * 16 + lo) * 32], hi);
        #pragma unroll
        for (int dt = 0; dt < 4; ++dt) {
            const _Float16* vr = vbase + (size_t)(dt * 16 + lo) * NSEQ + jj * 32 + hi * 16;
            v16h bv = *(const v16h*)vr;               // B fragment shared by both q sub-tiles
            #pragma unroll
            for (int qi = 0; qi < 2; ++qi)
                o[qi][dt] = wmma32(ap[qi], bv, o[qi][dt]);
        }
        __syncthreads();
    }

    // store merged-head output as f16 [B*N, E]
    #pragma unroll
    for (int qi = 0; qi < 2; ++qi)
        #pragma unroll
        for (int dt = 0; dt < 4; ++dt)
            #pragma unroll
            for (int r = 0; r < 8; ++r) {
                int row = q0 + qi * 16 + r + hi * 8;
                int col = h * DH + dt * 16 + lo;
                Oh[((size_t)b * NSEQ + row) * EDIM + col] = (_Float16)o[qi][dt][r];
            }
}

// ---------------- output projection: out = Oh @ W^T + b ----------------
// one wave per 16x16 output tile; K=1024 in chunks of 32
__global__ __launch_bounds__(32)
void proj_kernel(const _Float16* __restrict__ Oh, const _Float16* __restrict__ Wh,
                 const float* __restrict__ bias, float* __restrict__ out) {
    const int lane = threadIdx.x & 31;
    const int lo = lane & 15, hi = lane >> 4;
    const int r0 = blockIdx.x * 16;   // row tile over B*N = 4096
    const int e0 = blockIdx.y * 16;   // col tile over E

    const _Float16* arow = Oh + (size_t)(r0 + lo) * EDIM;
    const _Float16* brow = Wh + (size_t)(e0 + lo) * EDIM;   // W row-major [e][k] == B fragment layout

    v8f c = {};
    for (int kk = 0; kk < EDIM / 32; ++kk) {
        v16h a = load_a_frag(arow + kk * 32, hi);
        v16h bf = *(const v16h*)(brow + kk * 32 + hi * 16);
        c = wmma32(a, bf, c);
    }
    float bb = bias[e0 + lo];
    #pragma unroll
    for (int r = 0; r < 8; ++r)
        out[(size_t)(r0 + r + hi * 8) * EDIM + e0 + lo] = c[r] + bb;
}

extern "C" void kernel_launch(void* const* d_in, const int* in_sizes, int n_in,
                              void* d_out, int out_size, void* d_ws, size_t ws_size,
                              hipStream_t stream) {
    const float* q    = (const float*)d_in[0];
    const float* k    = (const float*)d_in[1];
    const float* v    = (const float*)d_in[2];
    const int*   mask = (const int*)d_in[3];
    const float* prev = (const float*)d_in[4];
    const float* W    = (const float*)d_in[5];
    const float* bvec = (const float*)d_in[6];

    char* ws = (char*)d_ws;
    const size_t szHeads = (size_t)BATCH * NH * NSEQ * DH * 2;   // 8 MB each
    _Float16* qh = (_Float16*)(ws);
    _Float16* kh = (_Float16*)(ws + szHeads);
    _Float16* vT = (_Float16*)(ws + 2 * szHeads);
    _Float16* Wh = (_Float16*)(ws + 3 * szHeads);
    _Float16* Oh = (_Float16*)(ws + 3 * szHeads + (size_t)EDIM * EDIM * 2);
    unsigned int* mbits = (unsigned int*)(ws + 3 * szHeads + (size_t)EDIM * EDIM * 2
                                             + (size_t)BATCH * NSEQ * EDIM * 2);

    const size_t totalQKV = (size_t)BATCH * NSEQ * EDIM;          // 4M
    cvt_qkv_kernel<<<(int)(totalQKV / 256), 256, 0, stream>>>(q, k, v, qh, kh, vT);
    cvt_w_kernel<<<(EDIM * EDIM) / 256, 256, 0, stream>>>(W, Wh);
    cvt_mask_kernel<<<(BATCH * NSEQ * (NSEQ / 32)) / 256, 256, 0, stream>>>(mask, mbits);

    attn_kernel<<<BATCH * NH * (NSEQ / 32), 32, 0, stream>>>(qh, kh, vT, mbits, prev, Oh);

    dim3 pg(BATCH * NSEQ / 16, EDIM / 16);
    proj_kernel<<<pg, 32, 0, stream>>>(Oh, Wh, bvec, (float*)d_out);
}
